// TemporalGraphNetwork_36618891166219
// MI455X (gfx1250) — compile-verified, matrix operand order checked by reading, not run on hardware
//
#include <hip/hip_runtime.h>
#include <hip/hip_bf16.h>
#include <math.h>
#include <stdint.h>

// ---------------------------------------------------------------------------
// Types for CDNA5 WMMA (wave32): 16x16x32 bf16 -> f32
// ---------------------------------------------------------------------------
typedef __attribute__((ext_vector_type(16))) __bf16 v16bf;
typedef __attribute__((ext_vector_type(8)))  float  v8f;

#define H_HEADS 4
#define CH      128     // channels per head
#define HID     512     // H_HEADS * CH

// ---------------------------------------------------------------------------
// Small device helpers
// ---------------------------------------------------------------------------
__device__ __forceinline__ unsigned short f32_to_bf16_rne(float f) {
  unsigned u = __float_as_uint(f);
  unsigned r = 0x7FFFu + ((u >> 16) & 1u);
  return (unsigned short)((u + r) >> 16);
}

// float atomic-max via sign-split int/uint atomics (init must be -inf bits)
__device__ __forceinline__ void atomicMaxF(float* addr, float v) {
  if (v >= 0.0f) atomicMax((int*)addr, __float_as_int(v));
  else           atomicMin((unsigned int*)addr, __float_as_uint(v));
}

// ---------------------------------------------------------------------------
// Fill / convert kernels
// ---------------------------------------------------------------------------
__global__ void fill_f32(float* __restrict__ p, float v, long n) {
  long i = (long)blockIdx.x * blockDim.x + threadIdx.x;
  if (i < n) p[i] = v;
}

__global__ void convert_bf16(const float* __restrict__ in,
                             unsigned short* __restrict__ out, long n) {
  long i = (long)blockIdx.x * blockDim.x + threadIdx.x;
  if (i < n) out[i] = f32_to_bf16_rne(in[i]);
}

// W (K x Nw, row-major f32)  ->  Wt (Nw x K, row-major bf16)
__global__ void convert_weight_t(const float* __restrict__ W,
                                 unsigned short* __restrict__ Wt, int K, int Nw) {
  long i = (long)blockIdx.x * blockDim.x + threadIdx.x;
  if (i >= (long)K * Nw) return;
  int n = (int)(i / K), k = (int)(i % K);
  Wt[i] = f32_to_bf16_rne(W[(long)k * Nw + n]);
}

// ---------------------------------------------------------------------------
// WMMA GEMM:  C[M x Nw] = A[M x K](bf16) * Bt[Nw x K](bf16)^T   (f32 accum)
// One wave per 16x16 C tile, K stepped by 32.
// ---------------------------------------------------------------------------
union Frag { uint4 u[2]; v16bf v; };

__global__ void wmma_gemm_bf16(const unsigned short* __restrict__ A,
                               const unsigned short* __restrict__ Bt,
                               float* __restrict__ C, int M, int K, int Nw) {
  const int lane   = threadIdx.x & 31;
  const int wave   = threadIdx.x >> 5;
  const int tilesN = Nw >> 4;
  const int tile   = blockIdx.x * (blockDim.x >> 5) + wave;
  const int total  = (M >> 4) * tilesN;
  if (tile >= total) return;
  const int tm  = tile / tilesN;
  const int tn  = tile % tilesN;
  const int row = lane & 15;        // M row (A) / N col (B)
  const int kg  = lane >> 4;        // which K-half this lane holds

  const unsigned short* pa = A  + (long)(tm * 16 + row) * K + kg * 8;
  const unsigned short* pb = Bt + (long)(tn * 16 + row) * K + kg * 8;

  v8f acc = {};
  for (int k0 = 0; k0 < K; k0 += 32) {
    Frag a, b;
    a.u[0] = *(const uint4*)(pa + k0);
    a.u[1] = *(const uint4*)(pa + k0 + 16);
    b.u[0] = *(const uint4*)(pb + k0);
    b.u[1] = *(const uint4*)(pb + k0 + 16);
    if (k0 + 32 < K) {                       // global_prefetch_b8 next K slab
      __builtin_prefetch(pa + k0 + 32, 0, 1);
      __builtin_prefetch(pb + k0 + 32, 0, 1);
    }
    acc = __builtin_amdgcn_wmma_f32_16x16x32_bf16(
        false, a.v, false, b.v, (short)0, acc, false, false);
  }

  float* pc = C + (long)(tm * 16 + (lane >> 4) * 8) * Nw + tn * 16 + (lane & 15);
#pragma unroll
  for (int v = 0; v < 8; ++v) pc[(long)v * Nw] = acc[v];
}

// ---------------------------------------------------------------------------
// Attention scores: al[n,h] = <xl[n,h,:], a_src[h,:]>, ar likewise.
// One wave per (node, head).
// ---------------------------------------------------------------------------
__global__ void attn_scores(const float* __restrict__ xl,
                            const float* __restrict__ a_s,
                            const float* __restrict__ a_d,
                            float* __restrict__ al, float* __restrict__ ar, int N) {
  int warp = (int)((blockIdx.x * (long)blockDim.x + threadIdx.x) >> 5);
  int lane = threadIdx.x & 31;
  if (warp >= N * H_HEADS) return;
  int n = warp / H_HEADS, h = warp % H_HEADS;
  const float* xr = xl + (long)n * HID + h * CH;
  const float* as = a_s + h * CH;
  const float* ad = a_d + h * CH;
  float sl = 0.f, sr = 0.f;
  for (int c = lane; c < CH; c += 32) { float xv = xr[c]; sl += xv * as[c]; sr += xv * ad[c]; }
#pragma unroll
  for (int off = 16; off; off >>= 1) {
    sl += __shfl_down(sl, off, 32);
    sr += __shfl_down(sr, off, 32);
  }
  if (lane == 0) { al[warp] = sl; ar[warp] = sr; }
}

// ---------------------------------------------------------------------------
// Edge pass 1: leaky-relu logits + segment-max over dst
// ---------------------------------------------------------------------------
__global__ void edge_logits(const int* __restrict__ ei, int E, int N,
                            const float* __restrict__ al, const float* __restrict__ ar,
                            float* __restrict__ elog, float* __restrict__ emax) {
  long idx = (long)blockIdx.x * blockDim.x + threadIdx.x;
  long tot = (long)(E + N) * H_HEADS;
  if (idx >= tot) return;
  int e = (int)(idx / H_HEADS), h = (int)(idx % H_HEADS);
  int s, d;
  if (e < E) { s = ei[e]; d = ei[E + e]; } else { s = d = e - E; }
  float v = al[s * H_HEADS + h] + ar[d * H_HEADS + h];
  v = v > 0.f ? v : 0.2f * v;
  elog[idx] = v;
  atomicMaxF(&emax[d * H_HEADS + h], v);
}

// ---------------------------------------------------------------------------
// Edge pass 2: exp(e - max) + segment-sum denominator
// ---------------------------------------------------------------------------
__global__ void edge_exp(const int* __restrict__ ei, int E, int N,
                         float* __restrict__ elog, const float* __restrict__ emax,
                         float* __restrict__ denom) {
  long idx = (long)blockIdx.x * blockDim.x + threadIdx.x;
  long tot = (long)(E + N) * H_HEADS;
  if (idx >= tot) return;
  int e = (int)(idx / H_HEADS), h = (int)(idx % H_HEADS);
  int d;
  if (e < E) { d = ei[E + e]; } else { d = e - E; }
  float ee = __expf(elog[idx] - emax[d * H_HEADS + h]);
  elog[idx] = ee;
  atomicAdd(&denom[d * H_HEADS + h], ee);
}

// ---------------------------------------------------------------------------
// Edge pass 3: out[dst, :] += alpha * xl[src, :]   (512-wide coalesced)
// ---------------------------------------------------------------------------
__global__ void edge_aggregate(const int* __restrict__ ei, int E, int N,
                               const float* __restrict__ xl,
                               const float* __restrict__ elog,
                               const float* __restrict__ denom,
                               float* __restrict__ outb) {
  __shared__ float alpha_sh[H_HEADS];
  int e = blockIdx.x;
  int s, d;
  if (e < E) { s = ei[e]; d = ei[E + e]; } else { s = d = e - E; }
  if (threadIdx.x < H_HEADS)
    alpha_sh[threadIdx.x] = elog[(long)e * H_HEADS + threadIdx.x] /
                            (denom[d * H_HEADS + threadIdx.x] + 1e-16f);
  __syncthreads();
  for (int c = threadIdx.x; c < HID; c += blockDim.x) {
    float a = alpha_sh[c >> 7];   // c / CH
    atomicAdd(&outb[(long)d * HID + c], a * xl[(long)s * HID + c]);
  }
}

// ---------------------------------------------------------------------------
// Post layers 1-2: h = relu(out + b) -> bf16 copy for next GEMM
// ---------------------------------------------------------------------------
__global__ void bias_relu_bf16(const float* __restrict__ outb,
                               const float* __restrict__ bias,
                               unsigned short* __restrict__ hb, long n, int C) {
  long i = (long)blockIdx.x * blockDim.x + threadIdx.x;
  if (i >= n) return;
  float v = outb[i] + bias[(int)(i % C)];
  hb[i] = f32_to_bf16_rne(v > 0.f ? v : 0.f);
}

// Post layer 3 (concat=False): mean over heads + bias + relu
__global__ void head_mean_bias_relu(const float* __restrict__ out3,
                                    const float* __restrict__ b3,
                                    float* __restrict__ h3, int N) {
  long i = (long)blockIdx.x * blockDim.x + threadIdx.x;
  if (i >= (long)N * CH) return;
  int n = (int)(i / CH), c = (int)(i % CH);
  float s = 0.f;
#pragma unroll
  for (int h = 0; h < H_HEADS; ++h) s += out3[(long)n * HID + h * CH + c];
  float v = s * (1.f / H_HEADS) + b3[c];
  h3[i] = v > 0.f ? v : 0.f;
}

// agg[src, :] += h3[dst, :]
__global__ void agg_scatter(const int* __restrict__ ei, int E, int N,
                            const float* __restrict__ h3, float* __restrict__ agg) {
  int e = blockIdx.x;
  int s, d;
  if (e < E) { s = ei[e]; d = ei[E + e]; } else { s = d = e - E; }
  for (int c = threadIdx.x; c < CH; c += blockDim.x)
    atomicAdd(&agg[(long)s * CH + c], h3[(long)d * CH + c]);
}

// ---------------------------------------------------------------------------
// Time MLP, folded: mean over edges of relu(t*w+b) -> 64-vector
// ---------------------------------------------------------------------------
__global__ void time_feat_sum(const float* __restrict__ t,
                              const float* __restrict__ tW1,
                              const float* __restrict__ tb1,
                              float* __restrict__ tsum, int E) {
  int j    = threadIdx.x & 63;
  int grp  = blockIdx.x * (blockDim.x >> 6) + (threadIdx.x >> 6);
  int ngrp = gridDim.x * (blockDim.x >> 6);
  float w = tW1[j], b = tb1[j], acc = 0.f;
  for (int e = grp; e < E; e += ngrp) {
    float v = t[e] * w + b;
    acc += v > 0.f ? v : 0.f;
  }
  atomicAdd(&tsum[j], acc);
}

__global__ void time_proj(const float* __restrict__ tsum,
                          const float* __restrict__ tW2,
                          const float* __restrict__ tb2,
                          float* __restrict__ tm, int E) {
  int j = threadIdx.x;          // 128 threads
  float acc = tb2[j];
  float inv = 1.f / (float)E;
#pragma unroll 8
  for (int k = 0; k < 64; ++k) acc += (tsum[k] * inv) * tW2[k * CH + j];
  tm[j] = acc;
}

// ---------------------------------------------------------------------------
// Fused tail: relu([agg,tm]@fW1+fb1) @ fW2 + fb2 -> LayerNorm -> + x@rW+rb
// One block of 128 threads per node.
// ---------------------------------------------------------------------------
__global__ void final_fused(const float* __restrict__ agg, const float* __restrict__ tm,
                            const float* __restrict__ x,
                            const float* __restrict__ fW1, const float* __restrict__ fb1,
                            const float* __restrict__ fW2, const float* __restrict__ fb2,
                            const float* __restrict__ ln_g, const float* __restrict__ ln_b,
                            const float* __restrict__ rW, const float* __restrict__ rb,
                            float* __restrict__ outp, int N) {
  __shared__ float zin[2 * CH];
  __shared__ float z1[CH];
  __shared__ float red[CH];
  int n = blockIdx.x, j = threadIdx.x;

  zin[j]      = agg[(long)n * CH + j];
  zin[CH + j] = tm[j];
  __syncthreads();

  float acc = fb1[j];
  for (int k = 0; k < 2 * CH; ++k) acc += zin[k] * fW1[k * CH + j];
  z1[j] = acc > 0.f ? acc : 0.f;
  __syncthreads();

  float z2 = fb2[j];
  for (int k = 0; k < CH; ++k) z2 += z1[k] * fW2[k * CH + j];

  float xr = rb[j];
  const float* xrow = x + (long)n * 64;
#pragma unroll 8
  for (int k = 0; k < 64; ++k) xr += xrow[k] * rW[k * CH + j];

  // LayerNorm over 128 channels
  red[j] = z2;
  __syncthreads();
  for (int s = CH / 2; s; s >>= 1) { if (j < s) red[j] += red[j + s]; __syncthreads(); }
  float mu = red[0] * (1.f / CH);
  __syncthreads();
  float dv = z2 - mu;
  red[j] = dv * dv;
  __syncthreads();
  for (int s = CH / 2; s; s >>= 1) { if (j < s) red[j] += red[j + s]; __syncthreads(); }
  float var = red[0] * (1.f / CH);
  float zn  = dv * rsqrtf(var + 1e-5f);
  outp[(long)n * CH + j] = zn * ln_g[j] + ln_b[j] + xr;
}

// ---------------------------------------------------------------------------
// Host orchestration
// ---------------------------------------------------------------------------
static void run_gat_layer(hipStream_t stream, int N, int E, int Etot, int Kin,
                          const int* ei, const float* W, const float* a_s, const float* a_d,
                          const unsigned short* xin_bf, unsigned short* wt,
                          float* xl, float* al, float* ar, float* emax, float* denom,
                          float* elog, float* outb) {
  const float NEG_INF = -1.0f / 0.0f;
  // weight -> bf16 transposed
  convert_weight_t<<<((long)Kin * HID + 255) / 256, 256, 0, stream>>>(W, wt, Kin, HID);
  // xl = x @ W  (WMMA bf16)
  int tiles  = (N / 16) * (HID / 16);
  int blocks = (tiles + 7) / 8;                     // 8 waves / 256-thread block
  wmma_gemm_bf16<<<blocks, 256, 0, stream>>>(xin_bf, wt, xl, N, Kin, HID);
  // attention scores
  attn_scores<<<((long)N * H_HEADS + 7) / 8, 256, 0, stream>>>(xl, a_s, a_d, al, ar, N);
  // init segment buffers
  fill_f32<<<((long)N * H_HEADS + 255) / 256, 256, 0, stream>>>(emax, NEG_INF, (long)N * H_HEADS);
  fill_f32<<<((long)N * H_HEADS + 255) / 256, 256, 0, stream>>>(denom, 0.f, (long)N * H_HEADS);
  fill_f32<<<((long)N * HID + 255) / 256, 256, 0, stream>>>(outb, 0.f, (long)N * HID);
  // softmax over incoming edges + weighted aggregation
  long et = (long)Etot * H_HEADS;
  edge_logits<<<(et + 255) / 256, 256, 0, stream>>>(ei, E, N, al, ar, elog, emax);
  edge_exp<<<(et + 255) / 256, 256, 0, stream>>>(ei, E, N, elog, emax, denom);
  edge_aggregate<<<Etot, 128, 0, stream>>>(ei, E, N, xl, elog, denom, outb);
}

extern "C" void kernel_launch(void* const* d_in, const int* in_sizes, int n_in,
                              void* d_out, int out_size, void* d_ws, size_t ws_size,
                              hipStream_t stream) {
  const float* x    = (const float*)d_in[0];
  const float* et   = (const float*)d_in[1];
  const float* W1   = (const float*)d_in[2];
  const float* as1  = (const float*)d_in[3];
  const float* ad1  = (const float*)d_in[4];
  const float* b1   = (const float*)d_in[5];
  const float* W2   = (const float*)d_in[6];
  const float* as2  = (const float*)d_in[7];
  const float* ad2  = (const float*)d_in[8];
  const float* b2   = (const float*)d_in[9];
  const float* W3   = (const float*)d_in[10];
  const float* as3  = (const float*)d_in[11];
  const float* ad3  = (const float*)d_in[12];
  const float* b3   = (const float*)d_in[13];
  const float* tW1  = (const float*)d_in[14];
  const float* tb1  = (const float*)d_in[15];
  const float* tW2  = (const float*)d_in[16];
  const float* tb2  = (const float*)d_in[17];
  const float* fW1  = (const float*)d_in[18];
  const float* fb1  = (const float*)d_in[19];
  const float* fb2p = (const float*)d_in[21];
  const float* fW2  = (const float*)d_in[20];
  const float* ln_g = (const float*)d_in[22];
  const float* ln_b = (const float*)d_in[23];
  const float* rW   = (const float*)d_in[24];
  const float* rb   = (const float*)d_in[25];
  const int*   ei   = (const int*)d_in[26];
  float*       out  = (float*)d_out;

  const int N    = in_sizes[0] / 64;   // 20000
  const int E    = in_sizes[1];        // 320000
  const int Etot = E + N;              // with self-loops

  // ---- workspace carve (256-byte aligned) ----
  size_t off = 0;
  char* base = (char*)d_ws;
  auto carve = [&](size_t bytes) -> void* {
    void* p = base + off;
    off += (bytes + 255) & ~(size_t)255;
    return p;
  };
  float*          xl    = (float*)carve((size_t)N * HID * 4);
  float*          outb  = (float*)carve((size_t)N * HID * 4);
  unsigned short* xb    = (unsigned short*)carve((size_t)N * HID * 2);
  unsigned short* wt    = (unsigned short*)carve((size_t)HID * HID * 2);
  float*          al    = (float*)carve((size_t)N * H_HEADS * 4);
  float*          ar    = (float*)carve((size_t)N * H_HEADS * 4);
  float*          emax  = (float*)carve((size_t)N * H_HEADS * 4);
  float*          denom = (float*)carve((size_t)N * H_HEADS * 4);
  float*          elog  = (float*)carve((size_t)Etot * H_HEADS * 4);
  float*          h3    = (float*)carve((size_t)N * CH * 4);
  float*          agg   = (float*)carve((size_t)N * CH * 4);
  float*          tsum  = (float*)carve(64 * 4);
  float*          tm    = (float*)carve(CH * 4);
  (void)ws_size;

  // ---- time embedding mean (algebraically folded) ----
  fill_f32<<<1, 64, 0, stream>>>(tsum, 0.f, 64);
  time_feat_sum<<<256, 256, 0, stream>>>(et, tW1, tb1, tsum, E);
  time_proj<<<1, CH, 0, stream>>>(tsum, tW2, tb2, tm, E);

  // ---- GAT layer 1 (in 64) ----
  convert_bf16<<<((long)N * 64 + 255) / 256, 256, 0, stream>>>(x, xb, (long)N * 64);
  run_gat_layer(stream, N, E, Etot, 64, ei, W1, as1, ad1, xb, wt,
                xl, al, ar, emax, denom, elog, outb);
  bias_relu_bf16<<<((long)N * HID + 255) / 256, 256, 0, stream>>>(outb, b1, xb, (long)N * HID, HID);

  // ---- GAT layer 2 (in 512) ----
  run_gat_layer(stream, N, E, Etot, HID, ei, W2, as2, ad2, xb, wt,
                xl, al, ar, emax, denom, elog, outb);
  bias_relu_bf16<<<((long)N * HID + 255) / 256, 256, 0, stream>>>(outb, b2, xb, (long)N * HID, HID);

  // ---- GAT layer 3 (in 512, concat=False) ----
  run_gat_layer(stream, N, E, Etot, HID, ei, W3, as3, ad3, xb, wt,
                xl, al, ar, emax, denom, elog, outb);
  head_mean_bias_relu<<<((long)N * CH + 255) / 256, 256, 0, stream>>>(outb, b3, h3, N);

  // ---- agg = segment_sum(h3[dst], src) ----
  fill_f32<<<((long)N * CH + 255) / 256, 256, 0, stream>>>(agg, 0.f, (long)N * CH);
  agg_scatter<<<Etot, 128, 0, stream>>>(ei, E, N, h3, agg);

  // ---- fused MLP + LayerNorm + residual ----
  final_fused<<<N, CH, 0, stream>>>(agg, tm, x, fW1, fb1, fW2, fb2p,
                                    ln_g, ln_b, rW, rb, out, N);
}
